// Sims_73770358276689
// MI455X (gfx1250) — matrix operand order repaired and checked by reading, not compiled, not run on hardware
//
#include <hip/hip_runtime.h>
#include <stdint.h>

// ---------------------------------------------------------------------------
// MI455X (gfx1250) fused pairwise-sqdist softmax
//   out[b,0,s,t] = softmax_t( (2*x[b,s]·x[b,t] - |x[b,s]|^2 - |x[b,t]|^2) / T )
// Strategy: split-bf16 GEMM (hi*hi + hi*lo + lo*hi) on V_WMMA_F32_16X16X32_BF16
// fused with row softmax; A-tile staged to LDS with the Tensor Data Mover.
// ---------------------------------------------------------------------------

typedef __bf16 bf16_t;
typedef __attribute__((ext_vector_type(16))) __bf16 v16bf;
typedef __attribute__((ext_vector_type(8)))  float  v8f;
typedef __attribute__((ext_vector_type(4)))  unsigned int u32x4;
typedef __attribute__((ext_vector_type(8)))  int  i32x8;
typedef __attribute__((ext_vector_type(4)))  int  i32x4;

#define S_DIM 2048
#define E_DIM 256
#define N_BATCH 16
#define TEMP_INV (1.0f / 13.544f)

// LDS layout (dynamic shared): A rows padded 256 -> 264 bf16 (+16B) so that the
// 16 lanes reading distinct rows of an A fragment hit distinct LDS banks.
#define A_LD      264                       // padded row stride in elements
#define A_BYTES   (32 * A_LD * 2)           // 16896 B per A tile (hi or lo)
#define SQ_OFF    (2 * A_BYTES)             // 33792
#define L_OFF     (SQ_OFF + S_DIM * 4)      // 41984
#define SMEM_BYTES (L_OFF + 32 * S_DIM * 4) // 304128 <= 320KB WGP LDS

__device__ __forceinline__ bf16_t f2bf(float f) {
    unsigned u = __builtin_bit_cast(unsigned, f);
    unsigned r = u + 0x7FFFu + ((u >> 16) & 1u);      // round-to-nearest-even
    unsigned short h = (unsigned short)(r >> 16);
    return __builtin_bit_cast(bf16_t, h);
}
__device__ __forceinline__ float bf2f(bf16_t h) {
    unsigned u = ((unsigned)__builtin_bit_cast(unsigned short, h)) << 16;
    return __builtin_bit_cast(float, u);
}

union FragU { uint4 q[2]; v16bf v; };

// A fragment 16x32 bf16 (ISA 7.12.2): lanes 0-15 row M=lane, K 0..7 (VGPR0-3)
// and K 16..23 (VGPR4-7); lanes 16-31 same rows, K 8..15 and 24..31.
__device__ __forceinline__ v16bf load_frag_a(const bf16_t* base, int row0,
                                             int kbase, int lane) {
    const int r = lane & 15, half = lane >> 4;
    const char* rp = (const char*)(base + (size_t)(row0 + r) * A_LD);
    FragU u;
    u.q[0] = *(const uint4*)(rp + (kbase + half * 8) * 2);
    u.q[1] = *(const uint4*)(rp + (kbase + 16 + half * 8) * 2);
    return u.v;
}

// B fragment 32x16 bf16: lanes 0-15 col N=lane, K 0..7,16..23; lanes 16-31 K 8..15,24..31.
// B[k,n] = x[t0+n, k] -> two 16B chunks 32B apart per lane of the row-major x panel.
__device__ __forceinline__ v16bf load_frag_b(const bf16_t* base, int tcol0,
                                             int kbase, int lane) {
    const int n = lane & 15, half = lane >> 4;
    const char* rp = (const char*)(base + (size_t)(tcol0 + n) * E_DIM);
    const char* p = rp + (kbase + half * 8) * 2;
    FragU u;
    u.q[0] = *(const uint4*)(p);
    u.q[1] = *(const uint4*)(p + 32);
    return u.v;
}

// ---------------------------------------------------------------------------
// Kernel 1: split x (fp32) into hi/lo bf16 and compute per-row |x|^2.
// One 32-lane wave per (b,s) row; 8 rows per 256-thread block.
// ---------------------------------------------------------------------------
__global__ void __launch_bounds__(256) sims_prep(const float* __restrict__ x,
                                                 bf16_t* __restrict__ hi,
                                                 bf16_t* __restrict__ lo,
                                                 float* __restrict__ sq) {
    const int lane = threadIdx.x & 31;
    const size_t row = (size_t)blockIdx.x * 8 + (threadIdx.x >> 5);
    const float* xr = x + row * E_DIM;
    bf16_t* hr = hi + row * E_DIM;
    bf16_t* lr = lo + row * E_DIM;
    float s = 0.0f;
#pragma unroll
    for (int j = 0; j < 8; ++j) {
        const int e = lane + j * 32;
        const float v = xr[e];
        const bf16_t h = f2bf(v);
        hr[e] = h;
        lr[e] = f2bf(v - bf2f(h));
        s += v * v;
    }
#pragma unroll
    for (int o = 16; o > 0; o >>= 1) s += __shfl_xor(s, o, 32);
    if (lane == 0) sq[row] = s;
}

// ---------------------------------------------------------------------------
// Kernel 2: per (batch, 32-row block): TDM-stage A tile (hi+lo) to LDS,
// split-bf16 WMMA gram over all 2048 columns into a 256KB LDS logit tile,
// then fused row softmax -> single global write.
// 8 waves: (wave&1) selects 16-row half, (wave>>1) selects 512-col strip.
// ---------------------------------------------------------------------------
__global__ void __launch_bounds__(256, 1)
sims_gemm_softmax(const bf16_t* __restrict__ hi, const bf16_t* __restrict__ lo,
                  const float* __restrict__ sq, float* __restrict__ out) {
    extern __shared__ char smem[];
    bf16_t* ldsAhi = (bf16_t*)smem;
    bf16_t* ldsAlo = (bf16_t*)(smem + A_BYTES);
    float* ldsSq = (float*)(smem + SQ_OFF);
    float* ldsL  = (float*)(smem + L_OFF);

    const int b = blockIdx.y;
    const int s0 = blockIdx.x * 32;
    const int tid = threadIdx.x;
    const int lane = tid & 31;
    const int wave = tid >> 5;

    const size_t batchOff = (size_t)b * S_DIM * E_DIM;
    const bf16_t* hiB = hi + batchOff;
    const bf16_t* loB = lo + batchOff;

    if (wave == 0) {
        // Tensor Data Mover: 1-D copy of 32x256 bf16 (16KB) with pad_enable:
        // pad_interval=6 (128 dwords = one 512B row) pad_amount=3 (+16B)
        // -> LDS row stride 528B = A_LD elements.
        // group1 word0 = data_size(2B)<<16 | pad_en | interval<<22 | amount<<25
        const i32x8 g1 = { (int)0x07910000,
                           0x20000000,        // tensor_dim0 = 8192 (lo16<<16)
                           0x00010000,        // tensor_dim0 hi | tensor_dim1=1
                           0x20000000,        // tile_dim0 = 8192
                           0, 8192, 0, 0 };   // tile_dim1/2=0, dim0_stride=8192
        const i32x4 g2 = { 1, 1, 0, 0 };          // tensor_dim2/3 = 1
        const i32x4 g3 = { 0, 0x00010000, 0, 0 }; // tensor_dim4 = 1
        const i32x8 gz = { 0, 0, 0, 0, 0, 0, 0, 0 }; // unused extra group
        const unsigned long long gaH =
            (unsigned long long)(uintptr_t)(hiB + (size_t)s0 * E_DIM);
        const unsigned long long gaL =
            (unsigned long long)(uintptr_t)(loB + (size_t)s0 * E_DIM);
        const u32x4 g0h = { 1u, 0u, (unsigned)gaH,
                            (unsigned)(gaH >> 32) | 0x80000000u }; // type=2
        const u32x4 g0l = { 1u, (unsigned)A_BYTES, (unsigned)gaL,
                            (unsigned)(gaL >> 32) | 0x80000000u };
        __builtin_amdgcn_tensor_load_to_lds(g0h, g1, g2, g3, gz, 0);
        __builtin_amdgcn_tensor_load_to_lds(g0l, g1, g2, g3, gz, 0);
        __builtin_amdgcn_s_wait_tensorcnt(0);
    }
    for (int i = tid; i < S_DIM; i += 256)
        ldsSq[i] = sq[(size_t)b * S_DIM + i];
    __syncthreads();

    // ---- gram tile: 32 rows x 2048 cols of logits into LDS ----
    const int mbase = (wave & 1) * 16;
    const int nbase = (wave >> 1) * 512;
    const int n = lane & 15, hv = lane >> 4;

    for (int tile = 0; tile < 32; ++tile) {
        const int colbase = nbase + tile * 16;
        v8f acc = {0.f, 0.f, 0.f, 0.f, 0.f, 0.f, 0.f, 0.f};
        for (int k = 0; k < E_DIM; k += 32) {
            const v16bf ah = load_frag_a(ldsAhi, mbase, k, lane);
            const v16bf al = load_frag_a(ldsAlo, mbase, k, lane);
            const v16bf bh = load_frag_b(hiB, colbase, k, lane);
            const v16bf bl = load_frag_b(loB, colbase, k, lane);
            // split-bf16: hi*hi + hi*lo + lo*hi ~= fp32 gram
            acc = __builtin_amdgcn_wmma_f32_16x16x32_bf16(
                false, ah, false, bh, (short)0, acc, false, false);
            acc = __builtin_amdgcn_wmma_f32_16x16x32_bf16(
                false, ah, false, bl, (short)0, acc, false, false);
            acc = __builtin_amdgcn_wmma_f32_16x16x32_bf16(
                false, al, false, bh, (short)0, acc, false, false);
        }
        // C/D layout: VGPR i -> rows (mbase+i, mbase+i+8), col = colbase+(lane&15)
        const int col = colbase + n;
        const float sqt = ldsSq[col];
#pragma unroll
        for (int i = 0; i < 8; ++i) {
            const int row = mbase + hv * 8 + i;
            const float sqs = ldsSq[s0 + row];
            ldsL[row * S_DIM + col] = (2.0f * acc[i] - sqs - sqt) * TEMP_INV;
        }
    }
    __syncthreads();

    // ---- fused row softmax: each wave owns 4 rows ----
    for (int rr = 0; rr < 4; ++rr) {
        const int row = wave * 4 + rr;
        float* L = ldsL + row * S_DIM;
        float m = -3.4e38f;
        for (int j = lane; j < S_DIM; j += 32) m = fmaxf(m, L[j]);
#pragma unroll
        for (int o = 16; o > 0; o >>= 1) m = fmaxf(m, __shfl_xor(m, o, 32));
        float s = 0.0f;
        for (int j = lane; j < S_DIM; j += 32) {
            const float e = __expf(L[j] - m);
            L[j] = e;
            s += e;
        }
#pragma unroll
        for (int o = 16; o > 0; o >>= 1) s += __shfl_xor(s, o, 32);
        const float inv = 1.0f / s;
        float* orow = out + ((size_t)b * S_DIM + (s0 + row)) * S_DIM;
        for (int j = lane; j < S_DIM; j += 32) orow[j] = L[j] * inv;
    }
}

extern "C" void kernel_launch(void* const* d_in, const int* in_sizes, int n_in,
                              void* d_out, int out_size, void* d_ws,
                              size_t ws_size, hipStream_t stream) {
    (void)in_sizes; (void)n_in; (void)out_size; (void)ws_size;
    const float* x = (const float*)d_in[0];
    float* out = (float*)d_out;

    // workspace carve: hi (16.78MB) | lo (16.78MB) | sq (128KB)
    const size_t elems = (size_t)N_BATCH * S_DIM * E_DIM;
    char* ws = (char*)d_ws;
    bf16_t* hi = (bf16_t*)ws;
    bf16_t* lo = (bf16_t*)(ws + elems * 2);
    float* sq = (float*)(ws + elems * 4);

    sims_prep<<<dim3((N_BATCH * S_DIM) / 8), dim3(256), 0, stream>>>(x, hi, lo, sq);

    (void)hipFuncSetAttribute((const void*)sims_gemm_softmax,
                              hipFuncAttributeMaxDynamicSharedMemorySize,
                              SMEM_BYTES);
    sims_gemm_softmax<<<dim3(S_DIM / 32, N_BATCH), dim3(256), SMEM_BYTES,
                        stream>>>(hi, lo, sq, out);
}
